// DAGAT_56659208569287
// MI455X (gfx1250) — matrix operand order repaired and checked by reading, not compiled, not run on hardware
//
#include <hip/hip_runtime.h>
#include <hip/hip_bf16.h>

// ---------------------------------------------------------------------------
// DAGAT forward for MI455X (gfx1250, wave32, WMMA bf16 16x16x32)
// All GEMM operands pre-converted to zero-padded bf16 so the WMMA inner loop
// has no conversion ALU and no bounds guards.
// ---------------------------------------------------------------------------

#define NROWS 8192
#define BM 128
#define BN 128
#define BK 32

typedef __bf16 v16bf __attribute__((ext_vector_type(16)));
typedef float  v8f   __attribute__((ext_vector_type(8)));

union Frag { v16bf v; unsigned u[8]; };

// fp32 -> bf16 bits, round to nearest even
__device__ __forceinline__ unsigned f2bf(float f) {
    unsigned u = __float_as_uint(f);
    return (u + 0x7FFFu + ((u >> 16) & 1u)) >> 16;
}
__device__ __forceinline__ unsigned pack_bf(float lo, float hi) {
    return f2bf(lo) | (f2bf(hi) << 16);
}

// 32-byte staging chunk kept strictly in VGPRs (no arrays / lambdas -> SROA ok)
struct U8v { uint4 lo, hi; };
__device__ __forceinline__ U8v ld8(const unsigned short* p) {
    U8v r;
    const uint4* q = (const uint4*)p;
    r.lo = q[0]; r.hi = q[1];
    return r;
}
__device__ __forceinline__ void st8(unsigned short* p, U8v v) {
    uint4* q = (uint4*)p;
    q[0] = v.lo; q[1] = v.hi;
}

// ---------------------------------------------------------------------------
// fp32 [M][K] -> bf16 [M][KP], zero-padded columns K..KP. One uint / thread.
// ---------------------------------------------------------------------------
__global__ __launch_bounds__(256) void cvt_bf16_pad(
    const float* __restrict__ X, unsigned short* __restrict__ Y,
    int M, int K, int KP)
{
    int idx = blockIdx.x * 256 + threadIdx.x;
    int rowu = KP >> 1;                 // uints per row
    int tot = M * rowu;
    if (idx >= tot) return;
    int m = idx / rowu;
    int j = (idx - m * rowu) << 1;      // even column
    float lo = (j     < K) ? X[(size_t)m * K + j]     : 0.f;
    float hi = (j + 1 < K) ? X[(size_t)m * K + j + 1] : 0.f;
    ((unsigned*)Y)[idx] = pack_bf(lo, hi);
}

// ---------------------------------------------------------------------------
// fp32 B[K][N] -> bf16 Bt[NP][KP] (transposed), zero-padded. 32x32 LDS tiles.
// grid = (KP/32, NP/32), block = 256.
// ---------------------------------------------------------------------------
__global__ __launch_bounds__(256) void cvt_transpose_pad(
    const float* __restrict__ B, unsigned short* __restrict__ Bt,
    int K, int N, int KP)
{
    __shared__ float sm[32][33];
    int k0 = blockIdx.x * 32, n0 = blockIdx.y * 32;
    int tx = threadIdx.x & 31, ty = threadIdx.x >> 5;   // 8 rows of 32
#pragma unroll
    for (int r = 0; r < 4; r++) {
        int kk = ty + 8 * r;
        float v = 0.f;
        if (k0 + kk < K && n0 + tx < N) v = B[(size_t)(k0 + kk) * N + n0 + tx];
        sm[kk][tx] = v;
    }
    __syncthreads();
#pragma unroll
    for (int r = 0; r < 4; r++) {
        int nn = ty + 8 * r;
        Bt[(size_t)(n0 + nn) * KP + k0 + tx] = (unsigned short)f2bf(sm[tx][nn]);
    }
}

// ---------------------------------------------------------------------------
// Tiled WMMA GEMM: C[M,N] = act( A @ B + bias )
//   A  : bf16 [M][KP]   (M multiple of 128, KP multiple of 32, zero-padded)
//   Bt : bf16 [>=ceil128(N)][KP]  (transposed, zero-padded)
// 256 threads = 8 waves in 4x2; each wave owns 32x64 (2x4 WMMA 16x16 tiles).
// Register-staged pipeline: tile i+1 global loads overlap WMMA of tile i.
// ACT: 0 none, 1 relu, 2 sigmoid, 3 elu
// ---------------------------------------------------------------------------
template <int ACT, bool BIAS>
__global__ __launch_bounds__(256) void gemm_bf16(
    const unsigned short* __restrict__ A,
    const unsigned short* __restrict__ Bt,
    const float* __restrict__ bias, float* __restrict__ C,
    int M, int N, int KP)
{
    __shared__ __attribute__((aligned(16))) unsigned short sA[BM * BK]; // [m][k]
    __shared__ __attribute__((aligned(16))) unsigned short sB[BN * BK]; // [n][k]

    const int tid   = threadIdx.x;
    const int lane  = tid & 31;
    const int wid   = tid >> 5;
    const int waveM = wid >> 1;      // 0..3
    const int waveN = wid & 1;       // 0..1
    const int hi    = lane >> 4;
    const int l16   = lane & 15;
    const int bm0   = blockIdx.y * BM;
    const int bn0   = blockIdx.x * BN;

    // staging: each thread owns 16 contiguous k of one row (A) / one col (Bt)
    const int sm_ = tid >> 1, skh = (tid & 1) << 4;

    const unsigned short* gA = A  + (size_t)(bm0 + sm_) * KP + skh;
    const unsigned short* gB = Bt + (size_t)(bn0 + sm_) * KP + skh;
    unsigned short* sAp = &sA[sm_ * BK + skh];
    unsigned short* sBp = &sB[sm_ * BK + skh];

    v8f acc[2][4];
#pragma unroll
    for (int i = 0; i < 2; i++)
#pragma unroll
        for (int j = 0; j < 4; j++)
#pragma unroll
            for (int e = 0; e < 8; e++) acc[i][j][e] = 0.f;

    // prologue: stage tile 0 into registers
    U8v ra = ld8(gA), rb = ld8(gB);

    for (int k0 = 0; k0 < KP; k0 += BK) {
        st8(sAp, ra);                  // regs (tile i) -> LDS
        st8(sBp, rb);
        __syncthreads();

        int k1 = k0 + BK;
        if (k1 < KP) {                 // overlap global loads with WMMA
            ra = ld8(gA + k1);
            rb = ld8(gB + k1);
        }

        // ---- per-lane fragment gathers (contiguous k = 16*hi + s, same order for A and B) ----
        Frag fa[2], fb[4];
#pragma unroll
        for (int sm = 0; sm < 2; sm++) {
            int m = waveM * 32 + sm * 16 + l16;
#pragma unroll
            for (int v = 0; v < 8; v++) {
                // A 16x32: lanes0-15 K 0..15; lanes16-31 K 16..31
                int k = (v << 1) + (hi << 4);
                fa[sm].u[v] = *(const unsigned*)&sA[m * BK + k];
            }
        }
#pragma unroll
        for (int sn = 0; sn < 4; sn++) {
            int n  = waveN * 64 + sn * 16 + l16;
            int kb = hi << 4;            // lanes16-31 hold K=16..31
#pragma unroll
            for (int v = 0; v < 8; v++)
                fb[sn].u[v] = *(const unsigned*)&sB[n * BK + kb + (v << 1)];
        }

#pragma unroll
        for (int sm = 0; sm < 2; sm++)
#pragma unroll
            for (int sn = 0; sn < 4; sn++)
                acc[sm][sn] = __builtin_amdgcn_wmma_f32_16x16x32_bf16(
                    false, fa[sm].v, false, fb[sn].v,
                    (short)0, acc[sm][sn], false, false);
        __syncthreads();
    }

    // ---- epilogue: C/D layout VGPR v -> M=v (lanes0-15) / v+8 (lanes16-31) --
#pragma unroll
    for (int sm = 0; sm < 2; sm++) {
#pragma unroll
        for (int sn = 0; sn < 4; sn++) {
            int n = bn0 + waveN * 64 + sn * 16 + l16;
            if (n >= N) continue;
            float bv = BIAS ? bias[n] : 0.f;
#pragma unroll
            for (int v = 0; v < 8; v++) {
                int m = bm0 + waveM * 32 + sm * 16 + v + (hi << 3);
                float x = acc[sm][sn][v] + bv;
                if (ACT == 1)      x = fmaxf(x, 0.f);
                else if (ACT == 2) x = 1.f / (1.f + __expf(-x));
                else if (ACT == 3) x = (x > 0.f) ? x : (__expf(x) - 1.f);
                C[(size_t)m * N + n] = x;
            }
        }
    }
}

// ---------------------------------------------------------------------------
// src[i] = Wh[i,:] . a[0:F] ; dst[i] = Wh[i,:] . a[F:2F]    (one block per row)
// ---------------------------------------------------------------------------
__global__ __launch_bounds__(256) void gat_attn_vec(
    const float* __restrict__ Wh, const float* __restrict__ a,
    int F, float* __restrict__ src, float* __restrict__ dst)
{
    int row = blockIdx.x;
    const float* w = Wh + (size_t)row * F;
    float s = 0.f, d = 0.f;
    for (int j = threadIdx.x; j < F; j += 256) {
        float v = w[j];
        s += v * a[j];
        d += v * a[F + j];
    }
    __shared__ float r0[256], r1[256];
    r0[threadIdx.x] = s; r1[threadIdx.x] = d;
    __syncthreads();
    for (int o = 128; o > 0; o >>= 1) {
        if (threadIdx.x < o) {
            r0[threadIdx.x] += r0[threadIdx.x + o];
            r1[threadIdx.x] += r1[threadIdx.x + o];
        }
        __syncthreads();
    }
    if (threadIdx.x == 0) { src[row] = r0[0]; dst[row] = r1[0]; }
}

// ---------------------------------------------------------------------------
// Masked-softmax attention row (leaky_relu(src_i + dst_j), adj mask) -> bf16
// One block per row; 3 passes (max / sum / write), recomputing e each pass.
// ---------------------------------------------------------------------------
__global__ __launch_bounds__(256) void attn_softmax(
    const float* __restrict__ adj, const float* __restrict__ src,
    const float* __restrict__ dst, unsigned short* __restrict__ att, int n)
{
    int row = blockIdx.x;
    const float* arow = adj + (size_t)row * n;
    float si = src[row];
    __shared__ float red[256];

    float mx = -3.0e38f;
    for (int j = threadIdx.x; j < n; j += 256) {
        if (arow[j] > 0.f) {
            float e = si + dst[j];
            e = (e > 0.f) ? e : 0.2f * e;
            mx = fmaxf(mx, e);
        }
    }
    red[threadIdx.x] = mx; __syncthreads();
    for (int o = 128; o > 0; o >>= 1) {
        if (threadIdx.x < o) red[threadIdx.x] = fmaxf(red[threadIdx.x], red[threadIdx.x + o]);
        __syncthreads();
    }
    mx = red[0]; __syncthreads();

    float sum = 0.f;
    for (int j = threadIdx.x; j < n; j += 256) {
        if (arow[j] > 0.f) {
            float e = si + dst[j];
            e = (e > 0.f) ? e : 0.2f * e;
            sum += __expf(e - mx);
        }
    }
    red[threadIdx.x] = sum; __syncthreads();
    for (int o = 128; o > 0; o >>= 1) {
        if (threadIdx.x < o) red[threadIdx.x] += red[threadIdx.x + o];
        __syncthreads();
    }
    float inv = 1.f / red[0];

    unsigned short* orow = att + (size_t)row * n;
    for (int j = threadIdx.x; j < n; j += 256) {
        unsigned short v = 0;
        if (arow[j] > 0.f) {
            float e = si + dst[j];
            e = (e > 0.f) ? e : 0.2f * e;
            v = (unsigned short)f2bf(__expf(e - mx) * inv);
        }
        orow[j] = v;
    }
}

// ---------------------------------------------------------------------------
// Misc elementwise kernels
// ---------------------------------------------------------------------------
__global__ void concat_cols(const float* __restrict__ A, const float* __restrict__ B,
                            float* __restrict__ Y, int M, int d)
{
    int idx = blockIdx.x * blockDim.x + threadIdx.x;
    int tot = M * 2 * d;
    if (idx < tot) {
        int r = idx / (2 * d), c = idx - r * 2 * d;
        Y[idx] = (c < d) ? A[(size_t)r * d + c] : B[(size_t)r * d + (c - d)];
    }
}

__global__ void axpby(const float* __restrict__ A, const float* __restrict__ B,
                      float ca, float cb, float* __restrict__ O, int n)
{
    int i = blockIdx.x * blockDim.x + threadIdx.x;
    if (i < n) O[i] = ca * A[i] + cb * B[i];
}

__global__ void wcombine(const float* __restrict__ A, const float* __restrict__ B,
                         const float* __restrict__ w, float* __restrict__ O, int n)
{
    int i = blockIdx.x * blockDim.x + threadIdx.x;
    if (i < n) O[i] = w[0] * A[i] + w[1] * B[i];
}

// att = softmax(s / T, axis=1) for 2 cols; w = mean(att, axis=0). One block.
__global__ __launch_bounds__(256) void fuse_weights(
    const float* __restrict__ S, int M, float invT, float* __restrict__ w)
{
    float s0 = 0.f, s1 = 0.f;
    for (int r = threadIdx.x; r < M; r += 256) {
        float a = S[2 * r] * invT, b = S[2 * r + 1] * invT;
        float m = fmaxf(a, b);
        float ea = __expf(a - m), eb = __expf(b - m);
        float inv = 1.f / (ea + eb);
        s0 += ea * inv; s1 += eb * inv;
    }
    __shared__ float r0[256], r1[256];
    r0[threadIdx.x] = s0; r1[threadIdx.x] = s1; __syncthreads();
    for (int o = 128; o > 0; o >>= 1) {
        if (threadIdx.x < o) {
            r0[threadIdx.x] += r0[threadIdx.x + o];
            r1[threadIdx.x] += r1[threadIdx.x + o];
        }
        __syncthreads();
    }
    if (threadIdx.x == 0) { w[0] = r0[0] / (float)M; w[1] = r1[0] / (float)M; }
}

// row softmax, C<=16 columns; thread per row
__global__ void softmax_rows(const float* __restrict__ H, float* __restrict__ O, int M, int C)
{
    int r = blockIdx.x * blockDim.x + threadIdx.x;
    if (r >= M) return;
    const float* h = H + (size_t)r * C;
    float m = -3.0e38f;
    for (int c = 0; c < C; c++) m = fmaxf(m, h[c]);
    float s = 0.f;
    for (int c = 0; c < C; c++) s += __expf(h[c] - m);
    float inv = 1.f / s;
    float* o = O + (size_t)r * C;
    for (int c = 0; c < C; c++) o[c] = __expf(h[c] - m) * inv;
}

// student-t cluster assignment, V=1 => q = (1+||z-mu||^2)^-1, normalized
__global__ void cluster_q(const float* __restrict__ Z, const float* __restrict__ cl,
                          float* __restrict__ Q, int M, int NC, int NZ)
{
    int r = blockIdx.x * blockDim.x + threadIdx.x;
    if (r >= M) return;
    float zv[16];
    for (int j = 0; j < NZ; j++) zv[j] = Z[(size_t)r * NZ + j];
    float qv[16], tot = 0.f;
    for (int c = 0; c < NC; c++) {
        float d = 0.f;
        for (int j = 0; j < NZ; j++) {
            float t = zv[j] - cl[c * NZ + j];
            d += t * t;
        }
        float q = 1.f / (1.f + d);
        qv[c] = q; tot += q;
    }
    float inv = 1.f / tot;
    for (int c = 0; c < NC; c++) Q[(size_t)r * NC + c] = qv[c] * inv;
}

// ---------------------------------------------------------------------------
// Host-side dispatch helpers
// ---------------------------------------------------------------------------
static inline int nelem(int d) { return NROWS * d; }
static inline int ru32(int x)  { return (x + 31)  & ~31;  }
static inline int ru128(int x) { return (x + 127) & ~127; }

static void launch_gemm_bf(const unsigned short* A, const unsigned short* Bt,
                           const float* bias, float* C,
                           int M, int N, int KP, int act, hipStream_t st)
{
    dim3 g((N + BN - 1) / BN, (M + BM - 1) / BM), b(256);
    if (bias) {
        switch (act) {
            case 0: gemm_bf16<0, true><<<g, b, 0, st>>>(A, Bt, bias, C, M, N, KP); break;
            case 1: gemm_bf16<1, true><<<g, b, 0, st>>>(A, Bt, bias, C, M, N, KP); break;
            case 2: gemm_bf16<2, true><<<g, b, 0, st>>>(A, Bt, bias, C, M, N, KP); break;
            default: gemm_bf16<3, true><<<g, b, 0, st>>>(A, Bt, bias, C, M, N, KP); break;
        }
    } else {
        switch (act) {
            case 0: gemm_bf16<0, false><<<g, b, 0, st>>>(A, Bt, bias, C, M, N, KP); break;
            case 1: gemm_bf16<1, false><<<g, b, 0, st>>>(A, Bt, bias, C, M, N, KP); break;
            case 2: gemm_bf16<2, false><<<g, b, 0, st>>>(A, Bt, bias, C, M, N, KP); break;
            default: gemm_bf16<3, false><<<g, b, 0, st>>>(A, Bt, bias, C, M, N, KP); break;
        }
    }
}

// Full fp32 GEMM: convert A (pad) + B (transpose+pad) to bf16, then WMMA GEMM.
static void gemm_f32(const float* A, const float* B, const float* bias, float* C,
                     int N, int K, int act, hipStream_t st,
                     unsigned short* Abf, unsigned short* Btf)
{
    int KP = ru32(K), NP = ru128(N);
    int totA = NROWS * (KP >> 1);
    cvt_bf16_pad<<<(totA + 255) / 256, 256, 0, st>>>(A, Abf, NROWS, K, KP);
    cvt_transpose_pad<<<dim3(KP / 32, NP / 32), 256, 0, st>>>(B, Btf, K, N, KP);
    launch_gemm_bf(Abf, Btf, bias, C, NROWS, N, KP, act, st);
}

// One full GAT layer: Wh = xin@W; src/dst; masked softmax; elu(att@Wh)
static void run_gat(const float* xin, int Fin, int Fout,
                    const float* W, const float* avec, const float* adj,
                    float* Wh, float* srcv, float* dstv, unsigned short* att,
                    float* hout, unsigned short* Abf, unsigned short* Btf,
                    hipStream_t st)
{
    gemm_f32(xin, W, nullptr, Wh, Fout, Fin, 0, st, Abf, Btf);
    gat_attn_vec<<<NROWS, 256, 0, st>>>(Wh, avec, Fout, srcv, dstv);
    attn_softmax<<<NROWS, 256, 0, st>>>(adj, srcv, dstv, att, NROWS);
    // Wh fp32 [8192][F] -> WhT bf16 [F128][8192]; att is already bf16 A.
    int NP = ru128(Fout);
    cvt_transpose_pad<<<dim3(NROWS / 32, NP / 32), 256, 0, st>>>(Wh, Btf, NROWS, Fout, NROWS);
    launch_gemm_bf(att, Btf, nullptr, hout, NROWS, Fout, NROWS, 3, st);
}

// Fuse gate: h <- w0*h + w1*k  (w from small MLP softmax mean), in place on h
static void run_fuse(float* h, const float* k, int d,
                     const float* w1, const float* b1, const float* w2, const float* b2,
                     const float* w3, const float* b3,
                     float* ybuf, float* f1, float* f2, float* sbuf, float* wbuf,
                     unsigned short* Abf, unsigned short* Btf, hipStream_t st)
{
    int tot = NROWS * 2 * d;
    concat_cols<<<(tot + 255) / 256, 256, 0, st>>>(h, k, ybuf, NROWS, d);
    gemm_f32(ybuf, w1, b1, f1, 500, 2 * d, 1, st, Abf, Btf);
    gemm_f32(f1, w2, b2, f2, 100, 500, 1, st, Abf, Btf);
    gemm_f32(f2, w3, b3, sbuf, 2, 100, 2, st, Abf, Btf);
    fuse_weights<<<1, 256, 0, st>>>(sbuf, NROWS, 0.1f /*1/T*/, wbuf);
    int n = NROWS * d;
    wcombine<<<(n + 255) / 256, 256, 0, st>>>(h, k, wbuf, h, n);
}

// ---------------------------------------------------------------------------
extern "C" void kernel_launch(void* const* d_in, const int* in_sizes, int n_in,
                              void* d_out, int out_size, void* d_ws, size_t ws_size,
                              hipStream_t stream)
{
    (void)in_sizes; (void)n_in; (void)out_size; (void)ws_size;

    // --- inputs (setup_inputs dict order, flattened) ---
    const float* x   = (const float*)d_in[0];       // [8192,1024]
    const float* adj = (const float*)d_in[1];       // [8192,8192]
    const float* ae_w[8]; const float* ae_b[8];
    for (int i = 0; i < 8; i++) { ae_w[i] = (const float*)d_in[2 + 2 * i];
                                  ae_b[i] = (const float*)d_in[3 + 2 * i]; }
    const float* gat_W[5]; const float* gat_a[5];
    for (int i = 0; i < 5; i++) { gat_W[i] = (const float*)d_in[18 + 2 * i];
                                  gat_a[i] = (const float*)d_in[19 + 2 * i]; }
    const float* fw[4][6];
    for (int i = 0; i < 4; i++)
        for (int j = 0; j < 6; j++) fw[i][j] = (const float*)d_in[28 + 6 * i + j];
    const float* cluster = (const float*)d_in[52];  // [10,10]

    // --- output layout: x_bar | q | predict | z ---
    float* out      = (float*)d_out;
    float* x_bar    = out;
    float* q_out    = out + (size_t)NROWS * 1024;
    float* pred_out = q_out + (size_t)NROWS * 10;
    float* z_out    = pred_out + (size_t)NROWS * 10;

    // --- workspace bump allocator ---
    char* wp = (char*)d_ws;
    auto alloc = [&](size_t bytes) -> char* {
        char* p = wp;
        wp += (bytes + 255) & ~(size_t)255;
        return p;
    };
    float* h1   = (float*)alloc((size_t)NROWS * 500  * 4);
    float* h2   = (float*)alloc((size_t)NROWS * 500  * 4);
    float* h3   = (float*)alloc((size_t)NROWS * 2000 * 4);
    float* zb   = (float*)alloc((size_t)NROWS * 10   * 4);
    float* dec1 = (float*)alloc((size_t)NROWS * 2000 * 4);
    float* dec2 = (float*)alloc((size_t)NROWS * 500  * 4);
    float* dec3 = (float*)alloc((size_t)NROWS * 500  * 4);
    float* Wh   = (float*)alloc((size_t)NROWS * 2000 * 4);
    float* srcv = (float*)alloc((size_t)NROWS * 4);
    float* dstv = (float*)alloc((size_t)NROWS * 4);
    unsigned short* att = (unsigned short*)alloc((size_t)NROWS * NROWS * 2);
    float* hbuf = (float*)alloc((size_t)NROWS * 2000 * 4);
    float* hin  = (float*)alloc((size_t)NROWS * 2000 * 4);
    float* ybuf = (float*)alloc((size_t)NROWS * 4000 * 4);
    float* f1   = (float*)alloc((size_t)NROWS * 500  * 4);
    float* f2   = (float*)alloc((size_t)NROWS * 100  * 4);
    float* sbuf = (float*)alloc((size_t)NROWS * 2    * 4);
    float* wbuf = (float*)alloc(2 * 4);
    // bf16 staging: A up to [8192][4000], Bt up to [2048][8192]
    unsigned short* Abf = (unsigned short*)alloc((size_t)NROWS * 4000 * 2);
    unsigned short* Btf = (unsigned short*)alloc((size_t)2048 * NROWS * 2);

    // ---------------- Autoencoder ----------------
    gemm_f32(x,    ae_w[0], ae_b[0], h1,     500, 1024, 1, stream, Abf, Btf);
    gemm_f32(h1,   ae_w[1], ae_b[1], h2,     500,  500, 1, stream, Abf, Btf);
    gemm_f32(h2,   ae_w[2], ae_b[2], h3,    2000,  500, 1, stream, Abf, Btf);
    gemm_f32(h3,   ae_w[3], ae_b[3], zb,      10, 2000, 0, stream, Abf, Btf);
    gemm_f32(zb,   ae_w[4], ae_b[4], dec1,  2000,   10, 1, stream, Abf, Btf);
    gemm_f32(dec1, ae_w[5], ae_b[5], dec2,   500, 2000, 1, stream, Abf, Btf);
    gemm_f32(dec2, ae_w[6], ae_b[6], dec3,   500,  500, 1, stream, Abf, Btf);
    gemm_f32(dec3, ae_w[7], ae_b[7], x_bar, 1024,  500, 0, stream, Abf, Btf);

    // ---------------- GAT / fuse stack (SIGMA = 0.5) ----------------
    // layer 0: 1024 -> 500
    run_gat(x, 1024, 500, gat_W[0], gat_a[0], adj, Wh, srcv, dstv, att, hbuf, Abf, Btf, stream);
    run_fuse(hbuf, h1, 500, fw[0][0], fw[0][1], fw[0][2], fw[0][3], fw[0][4], fw[0][5],
             ybuf, f1, f2, sbuf, wbuf, Abf, Btf, stream);
    axpby<<<(nelem(500) + 255) / 256, 256, 0, stream>>>(hbuf, h1, 0.5f, 0.5f, hin, nelem(500));

    // layer 1: 500 -> 500
    run_gat(hin, 500, 500, gat_W[1], gat_a[1], adj, Wh, srcv, dstv, att, hbuf, Abf, Btf, stream);
    run_fuse(hbuf, h2, 500, fw[1][0], fw[1][1], fw[1][2], fw[1][3], fw[1][4], fw[1][5],
             ybuf, f1, f2, sbuf, wbuf, Abf, Btf, stream);
    axpby<<<(nelem(500) + 255) / 256, 256, 0, stream>>>(hbuf, h2, 0.5f, 0.5f, hin, nelem(500));

    // layer 2: 500 -> 2000
    run_gat(hin, 500, 2000, gat_W[2], gat_a[2], adj, Wh, srcv, dstv, att, hbuf, Abf, Btf, stream);
    run_fuse(hbuf, h3, 2000, fw[2][0], fw[2][1], fw[2][2], fw[2][3], fw[2][4], fw[2][5],
             ybuf, f1, f2, sbuf, wbuf, Abf, Btf, stream);
    axpby<<<(nelem(2000) + 255) / 256, 256, 0, stream>>>(hbuf, h3, 0.5f, 0.5f, hin, nelem(2000));

    // layer 3: 2000 -> 10
    run_gat(hin, 2000, 10, gat_W[3], gat_a[3], adj, Wh, srcv, dstv, att, hbuf, Abf, Btf, stream);
    run_fuse(hbuf, zb, 10, fw[3][0], fw[3][1], fw[3][2], fw[3][3], fw[3][4], fw[3][5],
             ybuf, f1, f2, sbuf, wbuf, Abf, Btf, stream);
    axpby<<<(nelem(10) + 255) / 256, 256, 0, stream>>>(hbuf, zb, 0.5f, 0.5f, hin, nelem(10));

    // layer 4: 10 -> 10
    run_gat(hin, 10, 10, gat_W[4], gat_a[4], adj, Wh, srcv, dstv, att, hbuf, Abf, Btf, stream);

    // predict = softmax(h, axis=1)
    softmax_rows<<<(NROWS + 255) / 256, 256, 0, stream>>>(hbuf, pred_out, NROWS, 10);

    // student-t cluster assignment q
    cluster_q<<<(NROWS + 255) / 256, 256, 0, stream>>>(zb, cluster, q_out, NROWS, 10, 10);

    // z output
    (void)hipMemcpyAsync(z_out, zb, (size_t)NROWS * 10 * 4, hipMemcpyDeviceToDevice, stream);
}
